// AttentionNoCache_20134806684250
// MI455X (gfx1250) — matrix operand order repaired and checked by reading, not compiled, not run on hardware
//
#include <hip/hip_runtime.h>
#include <hip/hip_bf16.h>

// ---------------------------------------------------------------------------
// CDNA5 (gfx1250, wave32) causal multi-head attention forward pass.
//  - v_wmma_f32_16x16x32_bf16 for all matmuls
//  - TDM (tensor_load_to_lds + TENSORcnt) double-buffered B staging in GEMM
//  - ASYNCcnt (global_load_async_to_lds_b128) V staging in attention
// ---------------------------------------------------------------------------

#define T_SEQ   4096
#define C_MODEL 1024
#define N_HEADS 16
#define HEAD_D  64
#define QKV_N   (3 * C_MODEL)   // 3072

typedef __attribute__((ext_vector_type(16))) __bf16 v16bf;
typedef __attribute__((ext_vector_type(8)))  float  v8f;
typedef __attribute__((ext_vector_type(4)))  unsigned int uint4v;
typedef __attribute__((ext_vector_type(4)))  unsigned int su4;   // D# group 0
typedef __attribute__((ext_vector_type(8)))  unsigned int su8;   // D# group 1

union Frag {
    v16bf v;          // 16 x bf16 = 8 VGPRs, WMMA A/B operand
    uint4v q[2];
    unsigned int u[8];
};

__device__ __forceinline__ unsigned short f2bf(float x) {
    unsigned u = __float_as_uint(x);
    u += 0x7FFFu + ((u >> 16) & 1u);   // round-to-nearest-even
    return (unsigned short)(u >> 16);
}

// Async 16-byte global -> LDS copy (per-lane addresses), tracked by ASYNCcnt.
__device__ __forceinline__ void async_ld_b128(unsigned lds_off, const void* gptr) {
    asm volatile("global_load_async_to_lds_b128 %0, %1, off"
                 :: "v"(lds_off), "v"(gptr) : "memory");
}
__device__ __forceinline__ void wait_async0() {
    asm volatile("s_wait_asynccnt 0x0" ::: "memory");
}

// TDM: DMA a 2D tile (tile_d0 x tile_d1 elements of 2B, row stride
// stride_elems) from global into contiguous LDS. D# per ISA ch.8:
//  group0: {count=1 | is_restore=0, lds_addr, global_addr[56:0], type=2}
//  group1: {mask=0,data_size=1(2B), tensor_dim0/1, tile_dim0/1, dim0_stride}
// 2-operand form => VADDR2/VADDR3 NULL (2D tensor). Tracked by TENSORcnt.
__device__ __forceinline__ void tdm_load_2d(unsigned lds_addr, const void* gaddr,
                                            unsigned tile_d0, unsigned tile_d1,
                                            unsigned tensor_d0, unsigned tensor_d1,
                                            unsigned stride_elems) {
    unsigned long long ga = (unsigned long long)gaddr;
    su4 g0 = { 1u,                                    // count=1 (valid user D#)
               lds_addr,                               // LDS byte address
               (unsigned)ga,                           // global_addr[31:0]
               (unsigned)(ga >> 32) | (2u << 30) };    // [56:32] | type=2
    su8 g1 = { 0x00010000u,                            // data_size=1 -> 2 bytes
               (tensor_d0 & 0xFFFFu) << 16,            // tensor_dim0[15:0]
               (tensor_d0 >> 16) | ((tensor_d1 & 0xFFFFu) << 16),
               (tensor_d1 >> 16) | (tile_d0 << 16),    // tile_dim0
               tile_d1,                                // tile_dim1 (tile_dim2=0)
               stride_elems,                           // tensor_dim0_stride[31:0]
               0u, 0u };
    asm volatile("tensor_load_to_lds %0, %1" :: "s"(g0), "s"(g1) : "memory");
}

// A-matrix 16x32 bf16 fragment (this code's contiguous k order, same as B): lane = h*16 + t, row = t.
// v0..7 hold K = h*16 + {0..15}.
__device__ __forceinline__ void load_a_frag(Frag& f, const unsigned short* p,
                                            long elem_off, int h) {
    const unsigned int* d = (const unsigned int*)p + (elem_off >> 1);
    f.q[0] = *(const uint4v*)(d + h * 8);
    f.q[1] = *(const uint4v*)(d + h * 8 + 4);
}

// B-matrix 32x16 bf16 fragment from global: lane col = t, v_i holds K pair
// (h*16 + 2i, +1) -> dwords rowbase + h*8 + i.
__device__ __forceinline__ void load_b_frag(Frag& f, const unsigned short* p,
                                            long elem_off, int h) {
    const unsigned int* d = (const unsigned int*)p + (elem_off >> 1) + h * 8;
    f.q[0] = *(const uint4v*)(d);
    f.q[1] = *(const uint4v*)(d + 4);
}

// ---------------------------------------------------------------------------
// fp32 -> bf16 conversion
// ---------------------------------------------------------------------------
__global__ void cvt_f32_bf16(const float* __restrict__ in,
                             unsigned short* __restrict__ out, int n) {
    int i = blockIdx.x * blockDim.x + threadIdx.x;
    if (i < n) out[i] = f2bf(in[i]);
}

// ---------------------------------------------------------------------------
// C[M,N] = A[M,K] * Bsrc[N,K]^T   (A,Bsrc bf16 row-major; out bf16 or f32)
// grid = (N/128, M/128), block = 256 (8 waves). Wave w: rows m0+w*16..+15,
// all 128 cols (8 WMMA tiles). B tile DMA'd to LDS by wave 0 via TDM,
// double-buffered against the WMMA compute.
// ---------------------------------------------------------------------------
__global__ void gemm_bf16(const unsigned short* __restrict__ A,
                          const unsigned short* __restrict__ B,
                          unsigned short* __restrict__ Cb,
                          float* __restrict__ Cf,
                          int M, int N, int K, int outF32) {
    __shared__ unsigned short Bs[2][128 * 32];   // 2 x 8 KB: [col][k 0..31]

    const int tid  = threadIdx.x;
    const int w    = tid >> 5;
    const int lane = tid & 31;
    const int h    = lane >> 4;
    const int t    = lane & 15;
    const int m0 = blockIdx.y * 128 + w * 16;
    const int n0 = blockIdx.x * 128;

    v8f acc[8] = {};

    // ---- prologue: DMA first B tile into buffer 0 ----
    if (w == 0) {
        tdm_load_2d((unsigned)(size_t)&Bs[0][0], B + (long)n0 * K,
                    /*tile*/ 32, 128, /*tensor*/ (unsigned)K, (unsigned)N,
                    (unsigned)K);
        __builtin_amdgcn_s_wait_tensorcnt(0);
    }
    __syncthreads();

    for (int k0 = 0; k0 < K; k0 += 32) {
        const int cur = (k0 >> 5) & 1;

        // ---- overlap: wave 0 DMAs the next tile into the other buffer ----
        if (w == 0 && k0 + 32 < K)
            tdm_load_2d((unsigned)(size_t)&Bs[cur ^ 1][0],
                        B + (long)n0 * K + (k0 + 32),
                        32, 128, (unsigned)K, (unsigned)N, (unsigned)K);

        Frag a;
        load_a_frag(a, A, (long)(m0 + t) * K + k0, h);
        if (k0 + 32 < K)
            __builtin_prefetch(A + (long)(m0 + t) * K + k0 + 32, 0, 0);

        #pragma unroll
        for (int j = 0; j < 8; ++j) {
            Frag b;                               // col = 16j+t, dwords col*16 + h*8
            const unsigned int* pl =
                (const unsigned int*)&Bs[cur][0] + (16 * j + t) * 16 + h * 8;
            b.q[0] = *(const uint4v*)(pl);
            b.q[1] = *(const uint4v*)(pl + 4);
            acc[j] = __builtin_amdgcn_wmma_f32_16x16x32_bf16(
                false, a.v, false, b.v, (short)0, acc[j], false, false);
        }
        __syncthreads();                          // all waves done with Bs[cur]
        if (w == 0) __builtin_amdgcn_s_wait_tensorcnt(0);
        __syncthreads();                          // next buffer is ready
    }

    // C/D layout: v[i] -> row = i + 8*h, col = t.
    #pragma unroll
    for (int j = 0; j < 8; ++j) {
        #pragma unroll
        for (int i = 0; i < 8; ++i) {
            const int row = m0 + i + 8 * h;
            const int col = n0 + 16 * j + t;
            if (outF32) Cf[(long)row * N + col] = acc[j][i];
            else        Cb[(long)row * N + col] = f2bf(acc[j][i]);
        }
    }
}

// ---------------------------------------------------------------------------
// Flash attention (causal). qkv: bf16 [T][3*C], rows laid out [3][H][64].
// grid = (T/64, H), block = 128 (4 waves); wave w owns queries q0..q0+15.
// V tile (32 keys x 64 dims) staged per block via ASYNCcnt copies.
// ---------------------------------------------------------------------------
__global__ void attn_flash(const unsigned short* __restrict__ qkv,
                           unsigned short* __restrict__ attn_out) {
    __shared__ unsigned short Plds[4 * 16 * 32];   // per-wave 16x32 bf16 P tile
    __shared__ unsigned short Vs[32 * 64];         // shared V tile [key][dim]

    const int tid  = threadIdx.x;
    const int w    = tid >> 5;
    const int lane = tid & 31;
    const int h    = lane >> 4;
    const int t    = lane & 15;
    const int q0b  = blockIdx.x * 64;
    const int q0   = q0b + w * 16;
    const int head = blockIdx.y;

    const long qoff = (long)head * HEAD_D;
    const long koff = (long)C_MODEL + head * HEAD_D;
    const long voff = 2L * C_MODEL + head * HEAD_D;
    const float scale = 0.125f;                    // 1/sqrt(64)

    Frag aq[2];
    #pragma unroll
    for (int c = 0; c < 2; ++c)
        load_a_frag(aq[c], qkv, (long)(q0 + t) * QKV_N + qoff + c * 32, h);

    v8f   o[4] = {};
    float mrow[8], lrow[8];
    #pragma unroll
    for (int i = 0; i < 8; ++i) { mrow[i] = -1e30f; lrow[i] = 0.f; }

    unsigned short* pbase = &Plds[w * 16 * 32];

    for (int kb = 0; kb < q0b + 64; kb += 32) {
        // ---- cooperative V staging: 32 rows x 128B = 256 x 16B chunks ----
        #pragma unroll
        for (int ii = 0; ii < 2; ++ii) {
            const int idx = tid + 128 * ii;
            const int r = idx >> 3, c = idx & 7;
            const unsigned short* g = qkv + (long)(kb + r) * QKV_N + voff + c * 8;
            async_ld_b128((unsigned)(size_t)&Vs[r * 64 + c * 8], g);
        }
        wait_async0();
        __syncthreads();

        if (kb <= q0 + 15) {       // wave-uniform causal predicate
            // ---- scores: two 16-key subtiles ----
            v8f s[2];
            #pragma unroll
            for (int ssub = 0; ssub < 2; ++ssub) {
                const int ks = kb + 16 * ssub;
                v8f sc = {};
                #pragma unroll
                for (int c = 0; c < 2; ++c) {
                    Frag bk;
                    load_b_frag(bk, qkv, (long)(ks + t) * QKV_N + koff + c * 32, h);
                    sc = __builtin_amdgcn_wmma_f32_16x16x32_bf16(
                        false, aq[c].v, false, bk.v, (short)0, sc, false, false);
                }
                const int kg = ks + t;
                #pragma unroll
                for (int i = 0; i < 8; ++i) {
                    const int qg = q0 + i + 8 * h;
                    sc[i] = (kg <= qg) ? sc[i] * scale : -1e30f;
                }
                s[ssub] = sc;
            }

            // ---- online softmax (rows spread over 16 lanes of each half) ----
            #pragma unroll
            for (int i = 0; i < 8; ++i) {
                float smax = fmaxf(s[0][i], s[1][i]);
                #pragma unroll
                for (int m = 1; m < 16; m <<= 1)
                    smax = fmaxf(smax, __shfl_xor(smax, m, 32));
                const float mnew  = fmaxf(mrow[i], smax);
                const float alpha = __expf(mrow[i] - mnew);
                const float p0 = __expf(s[0][i] - mnew);
                const float p1 = __expf(s[1][i] - mnew);
                float rsum = p0 + p1;
                #pragma unroll
                for (int m = 1; m < 16; m <<= 1)
                    rsum += __shfl_xor(rsum, m, 32);
                lrow[i] = lrow[i] * alpha + rsum;
                mrow[i] = mnew;
                #pragma unroll
                for (int j = 0; j < 4; ++j) o[j][i] *= alpha;
                const int r = i + 8 * h;
                pbase[r * 32 + t]      = f2bf(p0);
                pbase[r * 32 + 16 + t] = f2bf(p1);
            }

            // ---- reload P as A-layout fragment (same-wave LDS RAW) ----
            Frag pa;
            {
                const unsigned int* pl = (const unsigned int*)pbase + t * 16;
                pa.q[0] = *(const uint4v*)(pl + h * 8);
                pa.q[1] = *(const uint4v*)(pl + h * 8 + 4);
            }

            // ---- O += P * V, V fragments gathered from shared LDS tile ----
            #pragma unroll
            for (int j = 0; j < 4; ++j) {
                Frag vb;
                #pragma unroll
                for (int i = 0; i < 8; ++i) {
                    const int kk = h * 16 + 2 * i;
                    const unsigned lo = Vs[kk * 64 + j * 16 + t];
                    const unsigned hi = Vs[(kk + 1) * 64 + j * 16 + t];
                    vb.u[i] = lo | (hi << 16);
                }
                o[j] = __builtin_amdgcn_wmma_f32_16x16x32_bf16(
                    false, pa.v, false, vb.v, (short)0, o[j], false, false);
            }
        }
        __syncthreads();
    }

    // ---- normalize and store bf16 to attn_out [T][C] ----
    #pragma unroll
    for (int j = 0; j < 4; ++j) {
        #pragma unroll
        for (int i = 0; i < 8; ++i) {
            const int row = q0 + i + 8 * h;
            const int col = head * HEAD_D + j * 16 + t;
            attn_out[(long)row * C_MODEL + col] = f2bf(o[j][i] / lrow[i]);
        }
    }
}

// ---------------------------------------------------------------------------
// Host-side launch
// ---------------------------------------------------------------------------
extern "C" void kernel_launch(void* const* d_in, const int* in_sizes, int n_in,
                              void* d_out, int out_size, void* d_ws, size_t ws_size,
                              hipStream_t stream) {
    const float* x     = (const float*)d_in[0];   // [1,4096,1024]
    const float* wqkv  = (const float*)d_in[1];   // [3072,1024]
    const float* wout  = (const float*)d_in[2];   // [1024,1024]
    float* out = (float*)d_out;                   // [1,4096,1024] f32

    unsigned short* ws     = (unsigned short*)d_ws;
    unsigned short* Xbf    = ws;                                   //  8 MiB
    unsigned short* Wqkvbf = Xbf    + (long)T_SEQ * C_MODEL;       //  6 MiB
    unsigned short* Woutbf = Wqkvbf + (long)QKV_N * C_MODEL;       //  2 MiB
    unsigned short* Qkvbf  = Woutbf + (long)C_MODEL * C_MODEL;     // 24 MiB
    unsigned short* Attnbf = Qkvbf  + (long)T_SEQ * QKV_N;         //  8 MiB

    const int nx = T_SEQ * C_MODEL, nq = QKV_N * C_MODEL, no = C_MODEL * C_MODEL;
    cvt_f32_bf16<<<(nx + 255) / 256, 256, 0, stream>>>(x, Xbf, nx);
    cvt_f32_bf16<<<(nq + 255) / 256, 256, 0, stream>>>(wqkv, Wqkvbf, nq);
    cvt_f32_bf16<<<(no + 255) / 256, 256, 0, stream>>>(wout, Woutbf, no);

    // QKV = X * Wqkv^T  -> bf16
    gemm_bf16<<<dim3(QKV_N / 128, T_SEQ / 128), 256, 0, stream>>>(
        Xbf, Wqkvbf, Qkvbf, nullptr, T_SEQ, QKV_N, C_MODEL, 0);

    // causal flash attention -> bf16 [T][C]
    attn_flash<<<dim3(T_SEQ / 64, N_HEADS), 128, 0, stream>>>(Qkvbf, Attnbf);

    // out = Attn * Wout^T -> f32
    gemm_bf16<<<dim3(C_MODEL / 128, T_SEQ / 128), 256, 0, stream>>>(
        Attnbf, Woutbf, nullptr, out, T_SEQ, C_MODEL, C_MODEL, 1);
}
